// TransformerRegressor_23192823399077
// MI455X (gfx1250) — matrix operand order repaired and checked by reading, not compiled, not run on hardware
//
#include <hip/hip_runtime.h>
#include <hip/hip_bf16.h>
#include <math.h>

// ---------------------------------------------------------------------------
// MI455X (gfx1250) implementation notes:
//  * ~134 GFLOP of GEMM per call vs ~150MB traffic -> compute bound; use the
//    matrix pipe: v_wmma_f32_16x16x32_bf16 (fp32 accumulate, wave32).
//  * GEMM: 64x64 block tile, 8 wave32, 2 WMMAs/wave/K-step (shared A frag),
//    double-buffered LDS software pipeline (load next slab -> WMMA current ->
//    store next slab -> 1 barrier). Ragged K/N edges peeled out of the hot
//    loop; edge staging is branch-free (clamped address + select, no execz).
//  * Requires M % 64 == 0 (true for all call sites: M = 4096 tokens).
//  * Flash attention keeps the 4M-entry score matrix in LDS/VGPRs.
//  * Explicit s_wait_dscnt for intra-wave cross-lane LDS dependencies.
// ---------------------------------------------------------------------------

#define LNUM   2
#define DMODEL 512
#define NH     8
#define DHEAD  64
#define MFF    2048
#define O0N    132
#define O1N    3
#define BN     2
#define TN     2048
#define TOK    (BN * TN)        // 4096 tokens
#define QKVW   (3 * NH * DHEAD) // 1536
#define CAT0W  (DMODEL + O0N)   // 644  (row stride 644*4B = 16B aligned)
#define CAT1W  (DMODEL + O1N)   // 515
#define CAT1LD 520              // padded stride so float4 rows stay aligned

typedef __attribute__((ext_vector_type(16))) __bf16 v16bf;
typedef __attribute__((ext_vector_type(8)))  __bf16 v8bf;
typedef __attribute__((ext_vector_type(2)))  __bf16 v2bf;
typedef __attribute__((ext_vector_type(8)))  float  v8f;

__device__ __forceinline__ v16bf join16(v8bf lo, v8bf hi) {
  v16bf r;
#pragma unroll
  for (int i = 0; i < 8; ++i) { r[i] = lo[i]; r[8 + i] = hi[i]; }
  return r;
}

__device__ __forceinline__ float gelu_exact(float v) {
  return 0.5f * v * (1.0f + erff(v * 0.70710678118654752f));
}

// ---------------------------------------------------------------------------
// LayerNorm over D=512, one block (256 threads) per token.
// ---------------------------------------------------------------------------
__global__ __launch_bounds__(256) void ln_kernel(const float* __restrict__ x,
                                                 const float* __restrict__ g,
                                                 const float* __restrict__ bta,
                                                 float* __restrict__ out) {
  __shared__ float sh1[256];
  __shared__ float sh2[256];
  const int tid = threadIdx.x;
  const size_t t = blockIdx.x;
  const float* xr = x + t * DMODEL;
  float v0 = xr[tid];
  float v1 = xr[tid + 256];
  sh1[tid] = v0 + v1;
  sh2[tid] = v0 * v0 + v1 * v1;
  __syncthreads();
  for (int st = 128; st > 0; st >>= 1) {
    if (tid < st) { sh1[tid] += sh1[tid + st]; sh2[tid] += sh2[tid + st]; }
    __syncthreads();
  }
  const float mu = sh1[0] * (1.0f / DMODEL);
  const float var = sh2[0] * (1.0f / DMODEL) - mu * mu;
  const float rstd = rsqrtf(var + 1e-5f);
  float* orow = out + t * DMODEL;
  orow[tid]       = (v0 - mu) * rstd * g[tid]       + bta[tid];
  orow[tid + 256] = (v1 - mu) * rstd * g[tid + 256] + bta[tid + 256];
}

// ---------------------------------------------------------------------------
// GEMM staging load helpers: branch-free at thread level. The fast/slow split
// is on uniform predicates only, so no execmask serialization in either path.
// ---------------------------------------------------------------------------
__device__ __forceinline__ float4 loadA_slab(const float* __restrict__ A,
                                             int lda, int K, int gm, int k0,
                                             int col4, bool fullK) {
  if (fullK)  // uniform: whole 32-wide slab in bounds
    return *(const float4*)(A + (size_t)gm * lda + k0 + col4);
  const float* ar = A + (size_t)gm * lda;
  float4 av;
#pragma unroll
  for (int j = 0; j < 4; ++j) {
    const int gk = k0 + col4 + j;
    const int gkc = gk < K ? gk : K - 1;   // clamp: load always legal
    const float v = ar[gkc];
    ((float*)&av)[j] = (gk < K) ? v : 0.0f;
  }
  return av;
}

__device__ __forceinline__ float4 loadW_slab(const float* __restrict__ W,
                                             int K, int N, int gk, int nbase,
                                             int nn4, bool fullTile) {
  if (fullTile)  // uniform: full N tile and full K slab
    return *(const float4*)(W + (size_t)gk * N + nbase + nn4);
  const int gkc = gk < K ? gk : K - 1;
  const float* wr = W + (size_t)gkc * N;
  float4 wv;
#pragma unroll
  for (int j = 0; j < 4; ++j) {
    const int gn = nbase + nn4 + j;
    const int gnc = gn < N ? gn : N - 1;
    const float v = wr[gnc];
    ((float*)&wv)[j] = (gk < K && gn < N) ? v : 0.0f;
  }
  return wv;
}

// ---------------------------------------------------------------------------
// Tiled WMMA GEMM: C[M,N] = act(A[M,K](lda) @ W[K,N] + bias) (+ residual)
// 256 threads = 8 wave32, block tile 64x64, K-step 32, double-buffered LDS.
// Each wave owns one 16-row m-tile and two 16-col n-tiles (shared A frag).
// ---------------------------------------------------------------------------
__global__ __launch_bounds__(256) void gemm_wmma_kernel(
    const float* __restrict__ A, int lda, const float* __restrict__ W,
    const float* __restrict__ bias, const float* __restrict__ residual,
    float* __restrict__ C, int Mtot, int K, int N, int act) {
  __shared__ __attribute__((aligned(16))) __bf16 As[2][64][40];  // [m][k]
  __shared__ __attribute__((aligned(16))) __bf16 Ws[2][64][40];  // [n][k] (T)

  const int tid  = threadIdx.x;
  const int lane = tid & 31;
  const int w    = tid >> 5;
  const int mi   = w >> 1;             // 0..3  (16-row tile)
  const int ni0  = (w & 1) * 32;       // 0 or 32 (two 16-col tiles)
  const int kh   = lane >> 4;
  const int l16  = lane & 15;
  const int mbase = blockIdx.y * 64;
  const int nbase = blockIdx.x * 64;
  const bool fullN = (nbase + 64 <= N);

  // staging coordinates (2 float4 per thread per tile)
  const int arow0 = tid >> 3;          // 0..31  (+32 for second half)
  const int acol4 = (tid & 7) * 4;     // 0..28
  const int wkk0  = tid >> 4;          // 0..15  (+16 for second half)
  const int wnn4  = (tid & 15) * 4;    // 0..60

  const int nsteps = (K + 31) >> 5;

  // ---- prologue: stage slab 0 into buffer 0
  {
    const bool fullK = (32 <= K);
    float4 a0 = loadA_slab(A, lda, K, mbase + arow0,      0, acol4, fullK);
    float4 a1 = loadA_slab(A, lda, K, mbase + arow0 + 32, 0, acol4, fullK);
    float4 w0 = loadW_slab(W, K, N, wkk0,      nbase, wnn4, fullN && fullK);
    float4 w1 = loadW_slab(W, K, N, wkk0 + 16, nbase, wnn4, fullN && fullK);
    v2bf p;
    p[0] = (__bf16)a0.x; p[1] = (__bf16)a0.y; *(v2bf*)&As[0][arow0][acol4] = p;
    p[0] = (__bf16)a0.z; p[1] = (__bf16)a0.w; *(v2bf*)&As[0][arow0][acol4 + 2] = p;
    p[0] = (__bf16)a1.x; p[1] = (__bf16)a1.y; *(v2bf*)&As[0][arow0 + 32][acol4] = p;
    p[0] = (__bf16)a1.z; p[1] = (__bf16)a1.w; *(v2bf*)&As[0][arow0 + 32][acol4 + 2] = p;
#pragma unroll
    for (int j = 0; j < 4; ++j) {
      Ws[0][wnn4 + j][wkk0]      = (__bf16)((float*)&w0)[j];
      Ws[0][wnn4 + j][wkk0 + 16] = (__bf16)((float*)&w1)[j];
    }
  }
  __syncthreads();

  v8f acc0 = {}, acc1 = {};
  const int arow = mi * 16 + l16;

  for (int s = 0; s < nsteps; ++s) {
    const int buf = s & 1;
    const bool havenext = (s + 1 < nsteps);

    // ---- phase 1: issue next slab's global loads (latency hides under WMMA)
    float4 a0n, a1n, w0n, w1n;
    if (havenext) {
      const int k0n = (s + 1) << 5;
      const bool fullK = (k0n + 32 <= K);
      a0n = loadA_slab(A, lda, K, mbase + arow0,      k0n, acol4, fullK);
      a1n = loadA_slab(A, lda, K, mbase + arow0 + 32, k0n, acol4, fullK);
      w0n = loadW_slab(W, K, N, k0n + wkk0,      nbase, wnn4, fullN && fullK);
      w1n = loadW_slab(W, K, N, k0n + wkk0 + 16, nbase, wnn4, fullN && fullK);
      if (s + 2 < nsteps) {  // hint the slab after next (global_prefetch_b8)
        const int k0p = (s + 2) << 5;
        __builtin_prefetch(A + (size_t)(mbase + arow0) * lda +
                               (k0p + acol4 < K ? k0p + acol4 : K - 1), 0, 1);
        const int gkp = k0p + wkk0;
        __builtin_prefetch(W + (size_t)(gkp < K ? gkp : K - 1) * N + nbase +
                               wnn4, 0, 1);
      }
    }

    // ---- phase 2: fragments from current buffer + 2 WMMAs
    {
      v16bf av = join16(*(const v8bf*)&As[buf][arow][16 * kh],
                        *(const v8bf*)&As[buf][arow][16 * kh + 8]);
      v16bf bv0 = join16(*(const v8bf*)&Ws[buf][ni0 + l16][16 * kh],
                         *(const v8bf*)&Ws[buf][ni0 + l16][16 * kh + 8]);
      v16bf bv1 = join16(*(const v8bf*)&Ws[buf][ni0 + 16 + l16][16 * kh],
                         *(const v8bf*)&Ws[buf][ni0 + 16 + l16][16 * kh + 8]);
      acc0 = __builtin_amdgcn_wmma_f32_16x16x32_bf16(false, av, false, bv0,
                                                     (short)0, acc0, false, false);
      acc1 = __builtin_amdgcn_wmma_f32_16x16x32_bf16(false, av, false, bv1,
                                                     (short)0, acc1, false, false);
    }

    // ---- phase 3: convert + store next slab into the other buffer
    if (havenext) {
      const int nb = buf ^ 1;
      v2bf p;
      p[0] = (__bf16)a0n.x; p[1] = (__bf16)a0n.y; *(v2bf*)&As[nb][arow0][acol4] = p;
      p[0] = (__bf16)a0n.z; p[1] = (__bf16)a0n.w; *(v2bf*)&As[nb][arow0][acol4 + 2] = p;
      p[0] = (__bf16)a1n.x; p[1] = (__bf16)a1n.y; *(v2bf*)&As[nb][arow0 + 32][acol4] = p;
      p[0] = (__bf16)a1n.z; p[1] = (__bf16)a1n.w; *(v2bf*)&As[nb][arow0 + 32][acol4 + 2] = p;
#pragma unroll
      for (int j = 0; j < 4; ++j) {
        Ws[nb][wnn4 + j][wkk0]      = (__bf16)((float*)&w0n)[j];
        Ws[nb][wnn4 + j][wkk0 + 16] = (__bf16)((float*)&w1n)[j];
      }
    }
    __syncthreads();
  }

  // ---- epilogue: bias + activation + residual, C-layout M = r + 8*kh
  const int m0 = mbase + mi * 16 + 8 * kh;
  const int n0 = nbase + ni0 + l16;
  const int n1 = n0 + 16;
#pragma unroll
  for (int r = 0; r < 8; ++r) {
    const int m = m0 + r;
    if (m < Mtot) {
      if (n0 < N) {
        float v = acc0[r];
        if (bias) v += bias[n0];
        if (act == 1) v = gelu_exact(v);
        if (residual) v += residual[(size_t)m * N + n0];
        C[(size_t)m * N + n0] = v;
      }
      if (n1 < N) {
        float v = acc1[r];
        if (bias) v += bias[n1];
        if (act == 1) v = gelu_exact(v);
        if (residual) v += residual[(size_t)m * N + n1];
        C[(size_t)m * N + n1] = v;
      }
    }
  }
}

// ---------------------------------------------------------------------------
// Flash attention: grid (T/64, H, B), 128 threads (4 wave32).
// Each wave owns one 16-row q tile; streams 32-key K/V blocks through LDS.
// q@k^T and p@v run on v_wmma_f32_16x16x32_bf16; online softmax in LDS.
// ---------------------------------------------------------------------------
__global__ __launch_bounds__(128) void attn_kernel(
    const float* __restrict__ qkv, const unsigned char* __restrict__ mask,
    float* __restrict__ out) {
  __shared__ __attribute__((aligned(16))) __bf16 kL[32][72];   // [key][dh]
  __shared__ __attribute__((aligned(16))) __bf16 vT[64][40];   // [dh][key]
  __shared__ float sS[4][16][33];                               // scores per wave
  __shared__ __attribute__((aligned(16))) __bf16 pP[4][16][40]; // probs per wave
  __shared__ float rmax[4][16], rsum[4][16], ralpha[4][16];

  const int tid  = threadIdx.x;
  const int lane = tid & 31;
  const int w    = tid >> 5;
  const int kh   = lane >> 4;
  const int l16  = lane & 15;
  const int b    = blockIdx.z;
  const int h    = blockIdx.y;
  const int qt0  = blockIdx.x * 64 + w * 16;
  const float scale = 0.125f;  // DH^-0.5

  // --- load q fragments (K = DH = 64 -> two 16x32 A fragments), pre-scaled
  const int qrow = qt0 + l16;
  const float* qp = qkv + ((size_t)(b * TN) + qrow) * QKVW + h * DHEAD;
  v16bf qa0, qa1;
  {
    float4 c0 = *(const float4*)(qp + 8 * kh);
    float4 c1 = *(const float4*)(qp + 8 * kh + 4);
    float4 c2 = *(const float4*)(qp + 16 + 8 * kh);
    float4 c3 = *(const float4*)(qp + 16 + 8 * kh + 4);
    float4 c4 = *(const float4*)(qp + 32 + 8 * kh);
    float4 c5 = *(const float4*)(qp + 32 + 8 * kh + 4);
    float4 c6 = *(const float4*)(qp + 48 + 8 * kh);
    float4 c7 = *(const float4*)(qp + 48 + 8 * kh + 4);
#pragma unroll
    for (int j = 0; j < 4; ++j) {
      qa0[j]      = (__bf16)(((float*)&c0)[j] * scale);
      qa0[4 + j]  = (__bf16)(((float*)&c1)[j] * scale);
      qa0[8 + j]  = (__bf16)(((float*)&c2)[j] * scale);
      qa0[12 + j] = (__bf16)(((float*)&c3)[j] * scale);
      qa1[j]      = (__bf16)(((float*)&c4)[j] * scale);
      qa1[4 + j]  = (__bf16)(((float*)&c5)[j] * scale);
      qa1[8 + j]  = (__bf16)(((float*)&c6)[j] * scale);
      qa1[12 + j] = (__bf16)(((float*)&c7)[j] * scale);
    }
  }
  v8f o0 = {}, o1 = {}, o2 = {}, o3 = {};
  if (lane < 16) { rmax[w][lane] = -3.0e38f; rsum[w][lane] = 0.0f; }

  for (int jb = 0; jb < TN; jb += 32) {
    // --- cooperative stage of K (row-major) and V (transposed) tiles
#pragma unroll
    for (int it = 0; it < 4; ++it) {
      const int linear = tid + 128 * it;   // 0..511 float4 slots
      const int key = linear >> 4;         // 0..31
      const int dh4 = (linear & 15) * 4;   // 0..60
      const float* base =
          qkv + ((size_t)(b * TN) + jb + key) * QKVW + h * DHEAD + dh4;
      float4 kv = *(const float4*)(base + NH * DHEAD);       // k block
      float4 vv = *(const float4*)(base + 2 * NH * DHEAD);   // v block
      v2bf kp0 = {(__bf16)kv.x, (__bf16)kv.y};
      v2bf kp1 = {(__bf16)kv.z, (__bf16)kv.w};
      *(v2bf*)&kL[key][dh4]     = kp0;
      *(v2bf*)&kL[key][dh4 + 2] = kp1;
      vT[dh4 + 0][key] = (__bf16)vv.x;
      vT[dh4 + 1][key] = (__bf16)vv.y;
      vT[dh4 + 2][key] = (__bf16)vv.z;
      vT[dh4 + 3][key] = (__bf16)vv.w;
    }
    __syncthreads();

    // --- s = q @ k^T for two 16-key sub-tiles (2 WMMAs each over DH=64)
#pragma unroll
    for (int sub = 0; sub < 2; ++sub) {
      const int keyl = sub * 16 + l16;
      v16bf kb0 = join16(*(const v8bf*)&kL[keyl][8 * kh],
                         *(const v8bf*)&kL[keyl][16 + 8 * kh]);
      v16bf kb1 = join16(*(const v8bf*)&kL[keyl][32 + 8 * kh],
                         *(const v8bf*)&kL[keyl][48 + 8 * kh]);
      v8f s = {};
      s = __builtin_amdgcn_wmma_f32_16x16x32_bf16(false, qa0, false, kb0,
                                                  (short)0, s, false, false);
      s = __builtin_amdgcn_wmma_f32_16x16x32_bf16(false, qa1, false, kb1,
                                                  (short)0, s, false, false);
#pragma unroll
      for (int r = 0; r < 8; ++r) sS[w][r + 8 * kh][sub * 16 + l16] = s[r];
    }
    asm volatile("s_wait_dscnt 0x0" ::: "memory");

    // --- online softmax: lanes 0..15 own one row each
    if (lane < 16) {
      const int row = lane;
      const float mold = rmax[w][row];
      float mnew = mold;
#pragma unroll 1
      for (int c = 0; c < 32; ++c) {
        float sv = sS[w][row][c];
        if (!mask[(size_t)b * TN + jb + c]) sv = -3.0e38f;
        sS[w][row][c] = sv;
        mnew = fmaxf(mnew, sv);
      }
      const float alpha = __expf(mold - mnew);
      float psum = 0.0f;
#pragma unroll 1
      for (int c = 0; c < 32; ++c) {
        const float p = __expf(sS[w][row][c] - mnew);
        pP[w][row][c] = (__bf16)p;
        psum += p;
      }
      rmax[w][row] = mnew;
      rsum[w][row] = rsum[w][row] * alpha + psum;
      ralpha[w][row] = alpha;
    }
    asm volatile("s_wait_dscnt 0x0" ::: "memory");

    // --- rescale accumulators, then o += p @ v (K = 32 keys, 4 dh tiles)
    v16bf pa = join16(*(const v8bf*)&pP[w][l16][16 * kh],
                      *(const v8bf*)&pP[w][l16][16 * kh + 8]);
#pragma unroll
    for (int r = 0; r < 8; ++r) {
      const float a = ralpha[w][r + 8 * kh];
      o0[r] *= a; o1[r] *= a; o2[r] *= a; o3[r] *= a;
    }
    v16bf vb0 = join16(*(const v8bf*)&vT[0  + l16][16 * kh],
                       *(const v8bf*)&vT[0  + l16][16 * kh + 8]);
    v16bf vb1 = join16(*(const v8bf*)&vT[16 + l16][16 * kh],
                       *(const v8bf*)&vT[16 + l16][16 * kh + 8]);
    v16bf vb2 = join16(*(const v8bf*)&vT[32 + l16][16 * kh],
                       *(const v8bf*)&vT[32 + l16][16 * kh + 8]);
    v16bf vb3 = join16(*(const v8bf*)&vT[48 + l16][16 * kh],
                       *(const v8bf*)&vT[48 + l16][16 * kh + 8]);
    o0 = __builtin_amdgcn_wmma_f32_16x16x32_bf16(false, pa, false, vb0, (short)0, o0, false, false);
    o1 = __builtin_amdgcn_wmma_f32_16x16x32_bf16(false, pa, false, vb1, (short)0, o1, false, false);
    o2 = __builtin_amdgcn_wmma_f32_16x16x32_bf16(false, pa, false, vb2, (short)0, o2, false, false);
    o3 = __builtin_amdgcn_wmma_f32_16x16x32_bf16(false, pa, false, vb3, (short)0, o3, false, false);
    __syncthreads();
  }

  // --- normalize by running sum and store [token][H*DH]
  asm volatile("s_wait_dscnt 0x0" ::: "memory");
#pragma unroll
  for (int r = 0; r < 8; ++r) {
    const int row = r + 8 * kh;
    const float inv = 1.0f / rsum[w][row];
    const int t = qt0 + row;
    float* op = out + ((size_t)(b * TN) + t) * (NH * DHEAD) + h * DHEAD;
    op[0  + l16] = o0[r] * inv;
    op[16 + l16] = o1[r] * inv;
    op[32 + l16] = o2[r] * inv;
    op[48 + l16] = o3[r] * inv;
  }
}

// ---------------------------------------------------------------------------
// Helpers: concat [h | y] per token (ldc row stride); flat copy.
// ---------------------------------------------------------------------------
__global__ void concat_kernel(const float* __restrict__ h,
                              const float* __restrict__ y,
                              float* __restrict__ cat, int Dh, int Oy,
                              int ldc) {
  const int Wd = Dh + Oy;
  const size_t total = (size_t)TOK * Wd;
  for (size_t i = (size_t)blockIdx.x * blockDim.x + threadIdx.x; i < total;
       i += (size_t)gridDim.x * blockDim.x) {
    const size_t t = i / Wd;
    const int c = (int)(i % Wd);
    cat[t * ldc + c] = (c < Dh) ? h[t * Dh + c] : y[t * Oy + (c - Dh)];
  }
}

__global__ void copy_kernel(const float* __restrict__ src,
                            float* __restrict__ dst, size_t n) {
  for (size_t i = (size_t)blockIdx.x * blockDim.x + threadIdx.x; i < n;
       i += (size_t)gridDim.x * blockDim.x)
    dst[i] = src[i];
}

// ---------------------------------------------------------------------------
extern "C" void kernel_launch(void* const* d_in, const int* in_sizes, int n_in,
                              void* d_out, int out_size, void* d_ws,
                              size_t ws_size, hipStream_t stream) {
  (void)in_sizes; (void)n_in; (void)out_size; (void)ws_size;

  const float* x_in  = (const float*)d_in[0];
  const float* init0 = (const float*)d_in[1];
  const float* init1 = (const float*)d_in[2];
  const unsigned char* mask = (const unsigned char*)d_in[3];
  const float* ln1_g = (const float*)d_in[4];
  const float* ln1_b = (const float*)d_in[5];
  const float* qkv_W = (const float*)d_in[6];
  const float* out_W = (const float*)d_in[7];
  const float* out_b = (const float*)d_in[8];
  const float* ln2_g = (const float*)d_in[9];
  const float* ln2_b = (const float*)d_in[10];
  const float* ff_W1 = (const float*)d_in[11];
  const float* ff_b1 = (const float*)d_in[12];
  const float* ff_W2 = (const float*)d_in[13];
  const float* ff_b2 = (const float*)d_in[14];
  const float* ln3_g = (const float*)d_in[15];
  const float* ln3_b = (const float*)d_in[16];
  const float* reg0_W1 = (const float*)d_in[17];
  const float* reg0_b1 = (const float*)d_in[18];
  const float* reg0_W2 = (const float*)d_in[19];
  const float* reg0_b2 = (const float*)d_in[20];
  const float* ln4_g = (const float*)d_in[21];
  const float* ln4_b = (const float*)d_in[22];
  const float* reg1_W1 = (const float*)d_in[23];
  const float* reg1_b1 = (const float*)d_in[24];
  const float* reg1_W2 = (const float*)d_in[25];
  const float* reg1_b2 = (const float*)d_in[26];

  // ---- workspace carve-up (floats)
  float* ws = (float*)d_ws;
  float* hbuf  = ws;                                   // 4096*512
  float* qkvb  = hbuf  + (size_t)TOK * DMODEL;         // 4096*1536
  float* attno = qkvb  + (size_t)TOK * QKVW;           // 4096*512
  float* xA    = attno + (size_t)TOK * DMODEL;         // 4096*512
  float* xB    = xA    + (size_t)TOK * DMODEL;         // 4096*512
  float* ffb   = xB    + (size_t)TOK * DMODEL;         // 4096*2048
  float* catb  = ffb   + (size_t)TOK * MFF;            // 4096*644 (>= 4096*520)
  float* y0b   = catb  + (size_t)TOK * CAT0W;          // 4096*132
  float* y1b   = y0b   + (size_t)TOK * O0N;            // 4096*3

  // ---- y0/y1 running state
  copy_kernel<<<256, 256, 0, stream>>>(init0, y0b, (size_t)TOK * O0N);
  copy_kernel<<<64, 256, 0, stream>>>(init1, y1b, (size_t)TOK * O1N);

  const dim3 gAttn(TN / 64, NH, BN);
  const float* xcur = x_in;

  for (int l = 0; l < LNUM; ++l) {
    const float* qkvW_l = qkv_W + (size_t)l * DMODEL * QKVW;
    const float* outW_l = out_W + (size_t)l * DMODEL * DMODEL;
    const float* outb_l = out_b + (size_t)l * DMODEL;
    const float* ffW1_l = ff_W1 + (size_t)l * DMODEL * MFF;
    const float* ffb1_l = ff_b1 + (size_t)l * MFF;
    const float* ffW2_l = ff_W2 + (size_t)l * MFF * DMODEL;
    const float* ffb2_l = ff_b2 + (size_t)l * DMODEL;
    const float* r0W1_l = reg0_W1 + (size_t)l * CAT0W * MFF;
    const float* r0b1_l = reg0_b1 + (size_t)l * MFF;
    const float* r0W2_l = reg0_W2 + (size_t)l * MFF * O0N;
    const float* r0b2_l = reg0_b2 + (size_t)l * O0N;
    const float* r1W1_l = reg1_W1 + (size_t)l * CAT1W * MFF;
    const float* r1b1_l = reg1_b1 + (size_t)l * MFF;
    const float* r1W2_l = reg1_W2 + (size_t)l * MFF * O1N;
    const float* r1b2_l = reg1_b2 + (size_t)l * O1N;

    // --- attention block
    ln_kernel<<<TOK, 256, 0, stream>>>(xcur, ln1_g + l * DMODEL,
                                       ln1_b + l * DMODEL, hbuf);
    gemm_wmma_kernel<<<dim3(QKVW / 64, TOK / 64), 256, 0, stream>>>(
        hbuf, DMODEL, qkvW_l, nullptr, nullptr, qkvb, TOK, DMODEL, QKVW, 0);
    attn_kernel<<<gAttn, 128, 0, stream>>>(qkvb, mask, attno);
    gemm_wmma_kernel<<<dim3(DMODEL / 64, TOK / 64), 256, 0, stream>>>(
        attno, DMODEL, outW_l, outb_l, xcur, xA, TOK, DMODEL, DMODEL, 0);

    // --- FFN block
    ln_kernel<<<TOK, 256, 0, stream>>>(xA, ln2_g + l * DMODEL,
                                       ln2_b + l * DMODEL, hbuf);
    gemm_wmma_kernel<<<dim3(MFF / 64, TOK / 64), 256, 0, stream>>>(
        hbuf, DMODEL, ffW1_l, ffb1_l, nullptr, ffb, TOK, DMODEL, MFF, 1);
    gemm_wmma_kernel<<<dim3(DMODEL / 64, TOK / 64), 256, 0, stream>>>(
        ffb, MFF, ffW2_l, ffb2_l, xA, xB, TOK, MFF, DMODEL, 0);

    // --- regressor head 0
    ln_kernel<<<TOK, 256, 0, stream>>>(xB, ln3_g + l * DMODEL,
                                       ln3_b + l * DMODEL, hbuf);
    concat_kernel<<<1024, 256, 0, stream>>>(hbuf, y0b, catb, DMODEL, O0N,
                                            CAT0W);
    gemm_wmma_kernel<<<dim3(MFF / 64, TOK / 64), 256, 0, stream>>>(
        catb, CAT0W, r0W1_l, r0b1_l, nullptr, ffb, TOK, CAT0W, MFF, 1);
    gemm_wmma_kernel<<<dim3((O0N + 63) / 64, TOK / 64), 256, 0, stream>>>(
        ffb, MFF, r0W2_l, r0b2_l, y0b, y0b, TOK, MFF, O0N, 0);

    // --- regressor head 1
    ln_kernel<<<TOK, 256, 0, stream>>>(xB, ln4_g + l * DMODEL,
                                       ln4_b + l * DMODEL, hbuf);
    concat_kernel<<<1024, 256, 0, stream>>>(hbuf, y1b, catb, DMODEL, O1N,
                                            CAT1LD);
    gemm_wmma_kernel<<<dim3(MFF / 64, TOK / 64), 256, 0, stream>>>(
        catb, CAT1LD, r1W1_l, r1b1_l, nullptr, ffb, TOK, CAT1W, MFF, 1);
    gemm_wmma_kernel<<<dim3(1, TOK / 64), 256, 0, stream>>>(
        ffb, MFF, r1W2_l, r1b2_l, y1b, y1b, TOK, MFF, O1N, 0);

    xcur = xB;
  }

  // ---- outputs: (y0, y1) flattened in order
  float* outp = (float*)d_out;
  copy_kernel<<<256, 256, 0, stream>>>(y0b, outp, (size_t)TOK * O0N);
  copy_kernel<<<64, 256, 0, stream>>>(y1b, outp + (size_t)TOK * O0N,
                                      (size_t)TOK * O1N);
}
